// SAModule_25804163514713
// MI455X (gfx1250) — matrix operand order repaired and checked by reading, not compiled, not run on hardware
//
#include <hip/hip_runtime.h>
#include <hip/hip_bf16.h>

// ---------------------------------------------------------------------------
// PointNet++ SAModule for MI455X (gfx1250, wave32, WMMA)
//   B=16 clouds, P=4096 pts, C=64 feats, S=1024 samples, K=32 neighbors
//   MLP: 67 -> 64 (relu) -> 64 (relu) -> 128, max over K neighbors
// ---------------------------------------------------------------------------

#define B_   16
#define P_   4096
#define C_   64
#define S_   1024
#define K_   32
#define CIN_ 67
#define K1PAD 96          // CIN padded to 3 * 32
#define H1_  64
#define H2_  64
#define OUT_ 128
#define NQ_  (B_ * S_)    // 16384 sampled points

typedef __attribute__((ext_vector_type(16))) __bf16 v16bf;
typedef __attribute__((ext_vector_type(8)))  __bf16 v8bf;
typedef __attribute__((ext_vector_type(8)))  float  v8f;

// ===========================================================================
// Kernel 1: farthest point sampling (one workgroup per cloud)
// ===========================================================================
__global__ __launch_bounds__(256) void fps_kernel(const float* __restrict__ pos,
                                                  float* __restrict__ posq,
                                                  float* __restrict__ batchq)
{
    __shared__ float cx[P_], cy[P_], cz[P_];
    __shared__ float bpos[3];
    __shared__ float wv[8];
    __shared__ int   wi[8];

    const int b   = blockIdx.x;
    const int tid = threadIdx.x;
    const int lane = tid & 31, w = tid >> 5;
    const float* cp = pos + (long)b * P_ * 3;

    for (int p = tid; p < P_; p += 256) {
        cx[p] = cp[p * 3 + 0];
        cy[p] = cp[p * 3 + 1];
        cz[p] = cp[p * 3 + 2];
    }
    __syncthreads();

    // 16 points per thread, pinned in registers
    float px[16], py[16], pz[16], dd[16];
#pragma unroll
    for (int t = 0; t < 16; ++t) {
        int p = tid * 16 + t;
        px[t] = cx[p]; py[t] = cy[p]; pz[t] = cz[p];
        dd[t] = 3.4e38f;
    }

    int cur = 0;  // meaningful on tid==0 only
    for (int i = 0; i < S_; ++i) {
        if (tid == 0) {
            int o = b * S_ + i;
            float sx = cx[cur], sy = cy[cur], sz = cz[cur];
            posq[o * 3 + 0] = sx;
            posq[o * 3 + 1] = sy;
            posq[o * 3 + 2] = sz;
            batchq[o] = (float)b;
            bpos[0] = sx; bpos[1] = sy; bpos[2] = sz;
        }
        __syncthreads();

        float fx = bpos[0], fy = bpos[1], fz = bpos[2];
        float bv = -1.0f; int bi = 0;
#pragma unroll
        for (int t = 0; t < 16; ++t) {
            float dx = px[t] - fx, dy = py[t] - fy, dz = pz[t] - fz;
            float d = dx * dx + dy * dy + dz * dz;
            dd[t] = fminf(dd[t], d);
            bool g = dd[t] > bv;          // strict > keeps lowest index on tie
            bi = g ? (tid * 16 + t) : bi;
            bv = g ? dd[t] : bv;
        }
        // wave32 argmax reduction
#pragma unroll
        for (int off = 16; off; off >>= 1) {
            float ov = __shfl_xor(bv, off);
            int   oi = __shfl_xor(bi, off);
            bool take = (ov > bv) || (ov == bv && oi < bi);
            bv = take ? ov : bv;
            bi = take ? oi : bi;
        }
        if (lane == 0) { wv[w] = bv; wi[w] = bi; }
        __syncthreads();

        if (tid == 0) {
            float mb = wv[0]; int mi = wi[0];
            for (int k = 1; k < 8; ++k)
                if (wv[k] > mb || (wv[k] == mb && wi[k] < mi)) { mb = wv[k]; mi = wi[k]; }
            cur = mi;
        }
    }
}

// ===========================================================================
// Kernel 2: kNN (one wave per query; pos stays in L2, 786 KB << 192 MB)
// ===========================================================================
__global__ __launch_bounds__(128) void knn_kernel(const float* __restrict__ pos,
                                                  const float* __restrict__ posq,
                                                  int* __restrict__ nbr)
{
    __shared__ float sd[4][P_ / 4];   // 4 waves x 1024 candidates
    __shared__ int   si[4][P_ / 4];

    const int lane = threadIdx.x & 31;
    const int w    = threadIdx.x >> 5;
    const int q    = blockIdx.x * 4 + w;
    const int b    = q >> 10;                 // q / S_
    const float* cp = pos + (long)b * P_ * 3;

    const float qx = posq[(long)q * 3 + 0];
    const float qy = posq[(long)q * 3 + 1];
    const float qz = posq[(long)q * 3 + 2];

    float dk[K_]; int ik[K_];
#pragma unroll
    for (int t = 0; t < K_; ++t) { dk[t] = 3.4e38f; ik[t] = -1; }

    for (int p = lane; p < P_; p += 32) {
        float dx = cp[p * 3 + 0] - qx;
        float dy = cp[p * 3 + 1] - qy;
        float dz = cp[p * 3 + 2] - qz;
        float d = dx * dx + dy * dy + dz * dz;
        if (d < dk[K_ - 1]) {
            dk[K_ - 1] = d; ik[K_ - 1] = p;
            // one bubble pass sinks the new element into the sorted list
#pragma unroll
            for (int j = K_ - 1; j > 0; --j) {
                bool sw = dk[j] < dk[j - 1];
                float ta = sw ? dk[j - 1] : dk[j];
                float tb = sw ? dk[j]     : dk[j - 1];
                int   ia = sw ? ik[j - 1] : ik[j];
                int   ib = sw ? ik[j]     : ik[j - 1];
                dk[j] = ta; dk[j - 1] = tb;
                ik[j] = ia; ik[j - 1] = ib;
            }
        }
    }

    // dump per-lane sorted lists to LDS
#pragma unroll
    for (int t = 0; t < K_; ++t) {
        sd[w][lane * K_ + t] = dk[t];
        si[w][lane * K_ + t] = ik[t];
    }
    __syncthreads();

    // 32 rounds of wave-wide extract-min over the 32 list heads
    int h = 0;
    for (int r = 0; r < K_; ++r) {
        float v  = sd[w][lane * K_ + h];
        int  pid = si[w][lane * K_ + h];
        int  wl  = lane;
#pragma unroll
        for (int off = 16; off; off >>= 1) {
            float ov  = __shfl_xor(v, off);
            int   opi = __shfl_xor(pid, off);
            int   owl = __shfl_xor(wl, off);
            bool take = (ov < v) || (ov == v && opi < pid);
            v   = take ? ov  : v;
            pid = take ? opi : pid;
            wl  = take ? owl : wl;
        }
        if (lane == 0) nbr[(long)q * K_ + r] = pid;
        if (lane == wl) ++h;
    }
}

// ===========================================================================
// Kernel 3: fused gather + 3-layer MLP (bf16 WMMA, f32 accum) + max-aggregate
// 128 threads = 4 waves, 2 queries (64 rows) per block. Each wave owns M=16.
// Weights live in LDS *transposed* ([N][K]) so every B fragment is one
// contiguous 32-byte vector load (2x ds_load_b128 per fragment).
// ===========================================================================
__device__ __forceinline__ v8f wmma_bf16(v16bf a, v16bf b, v8f c)
{
    return __builtin_amdgcn_wmma_f32_16x16x32_bf16(
        /*neg_a=*/false, a, /*neg_b=*/false, b,
        /*c_mod=*/(short)0, c, /*reuse_a=*/false, /*reuse_b=*/false);
}

__device__ __forceinline__ unsigned pack2bf(float a, float b)
{
    unsigned short ua = __builtin_bit_cast(unsigned short, (__bf16)a);
    unsigned short ub = __builtin_bit_cast(unsigned short, (__bf16)b);
    return (unsigned)ua | ((unsigned)ub << 16);
}

// A fragment 16x32 bf16: lane holds row m = lane&15,
// K in {kb..kb+15}, kb = 0 (lanes 0-15) / 16 (lanes 16-31).
// Two contiguous 16B chunks -> two ds_load_b128.
__device__ __forceinline__ v16bf ld_afrag(const __bf16* base, int ldk, int lane, int kc)
{
    int m  = lane & 15;
    int kb = (lane < 16) ? 0 : 16;
    const __bf16* r = base + m * ldk + kc * 32 + kb;
    v8bf lo = *(const v8bf*)(r);
    v8bf hi = *(const v8bf*)(r + 8);
    return __builtin_shufflevector(lo, hi,
        0, 1, 2, 3, 4, 5, 6, 7, 8, 9, 10, 11, 12, 13, 14, 15);
}

// B fragment 32x16 bf16 from transposed weights wt[N][kdim]:
// lane holds col n = nt*16 + (lane&15), K = kc*32 + kb + 0..15,
// kb = 0 (lanes 0-15) / 16 (lanes 16-31) -> one contiguous 32B vector.
__device__ __forceinline__ v16bf ld_bfrag(const __bf16* wt, int kdim, int lane, int kc, int nt)
{
    int n  = nt * 16 + (lane & 15);
    int kb = (lane < 16) ? 0 : 16;
    return *(const v16bf*)(wt + n * kdim + kc * 32 + kb);
}

__global__ __launch_bounds__(128) void pointconv_mlp(
    const float* __restrict__ x,   const float* __restrict__ pos,
    const float* __restrict__ posq, const int* __restrict__ nbr,
    const float* __restrict__ W1,  const float* __restrict__ b1,
    const float* __restrict__ W2,  const float* __restrict__ b2,
    const float* __restrict__ W3,  const float* __restrict__ b3,
    float* __restrict__ out)
{
    // 58368 bytes of LDS, manually laid out with reuse:
    //   sA  (12288 B) is reused as sH2 (8192 B) after layer 1 is consumed
    //   sH1 ( 8192 B) is reused as sCM (2048 B) after layer 2 is consumed
    __shared__ __align__(32) unsigned char smem[58368];
    __bf16* sW1t = (__bf16*)(smem);            // [64][96]  (N x K, transposed)
    __bf16* sW2t = (__bf16*)(smem + 12288);    // [64][64]
    __bf16* sW3t = (__bf16*)(smem + 20480);    // [128][64]
    float*  sb1  = (float*)(smem + 36864);     // [64]
    float*  sb2  = (float*)(smem + 37120);     // [64]
    float*  sb3  = (float*)(smem + 37376);     // [128]
    __bf16* sA   = (__bf16*)(smem + 37888);    // [64][96]
    __bf16* sH2  = (__bf16*)(smem + 37888);    // alias of sA, [64][64]
    __bf16* sH1  = (__bf16*)(smem + 50176);    // [64][64]
    float*  sCM  = (float*)(smem + 50176);     // alias of sH1, [4][128]

    const int tid  = threadIdx.x;
    const int lane = tid & 31;
    const int w    = tid >> 5;
    const int qbase = blockIdx.x * 2;

    // ---- stage weights transposed + bf16-packed (b32 LDS stores) ----
    {
        unsigned* u = (unsigned*)sW1t;                 // pid = n*48 + kp
        for (int pid = tid; pid < 64 * 48; pid += 128) {
            int n = pid / 48, k = (pid - n * 48) * 2;
            float a  = (k     < CIN_) ? W1[k * H1_ + n]       : 0.0f;
            float bb = (k + 1 < CIN_) ? W1[(k + 1) * H1_ + n] : 0.0f;
            u[pid] = pack2bf(a, bb);
        }
    }
    {
        unsigned* u = (unsigned*)sW2t;                 // pid = n*32 + kp
        for (int pid = tid; pid < 64 * 32; pid += 128) {
            int n = pid >> 5, k = (pid & 31) * 2;
            u[pid] = pack2bf(W2[k * H2_ + n], W2[(k + 1) * H2_ + n]);
        }
    }
    {
        unsigned* u = (unsigned*)sW3t;                 // pid = n*32 + kp
        for (int pid = tid; pid < 128 * 32; pid += 128) {
            int n = pid >> 5, k = (pid & 31) * 2;
            u[pid] = pack2bf(W3[k * OUT_ + n], W3[(k + 1) * OUT_ + n]);
        }
    }
    if (tid < 64)  { sb1[tid] = b1[tid]; sb2[tid] = b2[tid]; }
    if (tid < 128) sb3[tid] = b3[tid];

    // ---- gather A: 64 rows x 96 cols = concat[x_j, pos_j - pos_i, pad] ----
    {
        unsigned* u = (unsigned*)sA;                   // pid = row*48 + cp
        for (int pid = tid; pid < 64 * 48; pid += 128) {
            int row = pid / 48, col = (pid - row * 48) * 2;
            int q = qbase + (row >> 5), j = row & 31;
            int b = q >> 10;
            int p = nbr[(long)q * K_ + j];
            long pt = (long)b * P_ + p;
            float a, bb;
            if (col + 1 < C_) {                        // both from x
                a  = x[pt * C_ + col];
                bb = x[pt * C_ + col + 1];
            } else if (col < CIN_) {                   // rel-pos region
                a  = (col     < CIN_) ? pos[pt * 3 + (col - C_)]
                                        - posq[(long)q * 3 + (col - C_)] : 0.0f;
                bb = (col + 1 < CIN_) ? pos[pt * 3 + (col + 1 - C_)]
                                        - posq[(long)q * 3 + (col + 1 - C_)] : 0.0f;
            } else {
                a = 0.0f; bb = 0.0f;
            }
            u[pid] = pack2bf(a, bb);
        }
    }
    __syncthreads();

    const int rowBase = w * 16;
    const int nlane   = lane & 15;
    const int half8   = (lane < 16) ? 0 : 8;

    // ---- layer 1: [16x96] x [96x64] + b1, relu -> sH1 ----
    {
        v16bf a0 = ld_afrag(sA + rowBase * K1PAD, K1PAD, lane, 0);
        v16bf a1 = ld_afrag(sA + rowBase * K1PAD, K1PAD, lane, 1);
        v16bf a2 = ld_afrag(sA + rowBase * K1PAD, K1PAD, lane, 2);
#pragma unroll
        for (int nt = 0; nt < 4; ++nt) {
            float bias = sb1[nt * 16 + nlane];
            v8f acc;
#pragma unroll
            for (int r = 0; r < 8; ++r) acc[r] = bias;
            acc = wmma_bf16(a0, ld_bfrag(sW1t, K1PAD, lane, 0, nt), acc);
            acc = wmma_bf16(a1, ld_bfrag(sW1t, K1PAD, lane, 1, nt), acc);
            acc = wmma_bf16(a2, ld_bfrag(sW1t, K1PAD, lane, 2, nt), acc);
#pragma unroll
            for (int r = 0; r < 8; ++r)
                sH1[(rowBase + half8 + r) * H1_ + nt * 16 + nlane] =
                    (__bf16)fmaxf(acc[r], 0.0f);
        }
    }
    __syncthreads();

    // ---- layer 2: [16x64] x [64x64] + b2, relu -> sH2 (aliases sA) ----
    {
        v16bf a0 = ld_afrag(sH1 + rowBase * H1_, H1_, lane, 0);
        v16bf a1 = ld_afrag(sH1 + rowBase * H1_, H1_, lane, 1);
#pragma unroll
        for (int nt = 0; nt < 4; ++nt) {
            float bias = sb2[nt * 16 + nlane];
            v8f acc;
#pragma unroll
            for (int r = 0; r < 8; ++r) acc[r] = bias;
            acc = wmma_bf16(a0, ld_bfrag(sW2t, H2_, lane, 0, nt), acc);
            acc = wmma_bf16(a1, ld_bfrag(sW2t, H2_, lane, 1, nt), acc);
#pragma unroll
            for (int r = 0; r < 8; ++r)
                sH2[(rowBase + half8 + r) * H2_ + nt * 16 + nlane] =
                    (__bf16)fmaxf(acc[r], 0.0f);
        }
    }
    __syncthreads();

    // ---- layer 3: [16x64] x [64x128] + b3, then column-max of each D tile ----
    {
        v16bf a0 = ld_afrag(sH2 + rowBase * H2_, H2_, lane, 0);
        v16bf a1 = ld_afrag(sH2 + rowBase * H2_, H2_, lane, 1);
#pragma unroll
        for (int nt = 0; nt < 8; ++nt) {
            float bias = sb3[nt * 16 + nlane];
            v8f acc;
#pragma unroll
            for (int r = 0; r < 8; ++r) acc[r] = bias;
            acc = wmma_bf16(a0, ld_bfrag(sW3t, H2_, lane, 0, nt), acc);
            acc = wmma_bf16(a1, ld_bfrag(sW3t, H2_, lane, 1, nt), acc);
            // D layout: lane holds rows half8..half8+7 of column nlane
            float pm = acc[0];
#pragma unroll
            for (int r = 1; r < 8; ++r) pm = fmaxf(pm, acc[r]);
            pm = fmaxf(pm, __shfl_xor(pm, 16));   // combine the two row halves
            if (lane < 16) sCM[w * OUT_ + nt * 16 + lane] = pm;
        }
    }
    __syncthreads();

    // ---- combine the two 16-row tiles of each query, write output ----
    for (int e = tid; e < 2 * OUT_; e += 128) {
        int ql = e >> 7, col = e & 127;
        float v = fmaxf(sCM[(2 * ql) * OUT_ + col], sCM[(2 * ql + 1) * OUT_ + col]);
        out[(long)(qbase + ql) * OUT_ + col] = v;
    }
}

// ===========================================================================
// Host-side launcher
// ===========================================================================
extern "C" void kernel_launch(void* const* d_in, const int* in_sizes, int n_in,
                              void* d_out, int out_size, void* d_ws, size_t ws_size,
                              hipStream_t stream)
{
    const float* x    = (const float*)d_in[0];
    const float* pos  = (const float*)d_in[1];
    // d_in[2] = batch (unused; layout is static contiguous segments)
    const float* W1   = (const float*)d_in[3];
    const float* b1   = (const float*)d_in[4];
    const float* W2   = (const float*)d_in[5];
    const float* b2   = (const float*)d_in[6];
    const float* W3   = (const float*)d_in[7];
    const float* b3   = (const float*)d_in[8];

    float* out    = (float*)d_out;                 // [NQ, 128]
    float* posq   = out + (long)NQ_ * OUT_;        // [NQ, 3]
    float* batchq = posq + (long)NQ_ * 3;          // [NQ]
    int*   nbr    = (int*)d_ws;                    // [NQ, K] = 2 MiB scratch

    fps_kernel<<<B_, 256, 0, stream>>>(pos, posq, batchq);
    knn_kernel<<<NQ_ / 4, 128, 0, stream>>>(pos, posq, nbr);
    pointconv_mlp<<<NQ_ / 2, 128, 0, stream>>>(x, pos, posq, nbr,
                                               W1, b1, W2, b2, W3, b3, out);
}
